// AEGAN_68925635166342
// MI455X (gfx1250) — matrix operand order repaired and checked by reading, not compiled, not run on hardware
//
#include <hip/hip_runtime.h>
#include <math.h>

// ---------------------------------------------------------------------------
// AEGAN forward for MI455X (gfx1250, wave32). All GEMM-shaped work goes
// through v_wmma_f32_16x16x32_f16 (f16 in, f32 accumulate), 2x2 register
// blocked (32x32 tile per wave -> 4 WMMAs per K-step, 2x fragment reuse).
// Bandwidth-dominant intermediates (attention probs, FNN1 hidden, V in two
// transposed layouts) are stored f16 so the big GEMMs stream pure b128 loads.
// ---------------------------------------------------------------------------

typedef __attribute__((ext_vector_type(16))) _Float16 v16h;
typedef __attribute__((ext_vector_type(8)))  _Float16 h8;
typedef __attribute__((ext_vector_type(8)))  float    v8f;

constexpr int  NB  = 2;       // batch
constexpr int  NN_ = 256;     // N
constexpr int  FD  = 64;      // EF == NF
constexpr int  NH  = 4;       // heads
constexpr long ME  = (long)NB * NN_ * NN_;   // 131072 edge rows

#define DEVFN __device__ __forceinline__

DEVFN float wsum(float v) {
  #pragma unroll
  for (int o = 16; o > 0; o >>= 1) v += __shfl_xor(v, o, 32);
  return v;
}
DEVFN float wmax(float v) {
  #pragma unroll
  for (int o = 16; o > 0; o >>= 1) v = fmaxf(v, __shfl_xor(v, o, 32));
  return v;
}
DEVFN float lrelu(float v) { return v > 0.f ? v : 0.2f * v; }
DEVFN float sigm(float v)  { return 1.f / (1.f + __expf(-v)); }

// ---- A fragment loaders (unit k-stride) ------------------------------------
// Lane layout: row = rowBase+lane%16; K run [kb,kb+16),
// kb = kk + (lane/16)*16.
DEVFN v16h fragA_any(const float* __restrict__ p)
{
  v16h a;
  const float4 x0 = *(const float4*)(p);
  const float4 x1 = *(const float4*)(p + 4);
  const float4 x2 = *(const float4*)(p + 8);
  const float4 x3 = *(const float4*)(p + 12);
  a[0]=(_Float16)x0.x; a[1]=(_Float16)x0.y; a[2]=(_Float16)x0.z; a[3]=(_Float16)x0.w;
  a[4]=(_Float16)x1.x; a[5]=(_Float16)x1.y; a[6]=(_Float16)x1.z; a[7]=(_Float16)x1.w;
  a[8]=(_Float16)x2.x; a[9]=(_Float16)x2.y; a[10]=(_Float16)x2.z; a[11]=(_Float16)x2.w;
  a[12]=(_Float16)x3.x; a[13]=(_Float16)x3.y; a[14]=(_Float16)x3.z; a[15]=(_Float16)x3.w;
  return a;
}
DEVFN v16h fragA_any(const _Float16* __restrict__ p)
{
  v16h a;
  const h8 r0 = *(const h8*)(p);
  const h8 r1 = *(const h8*)(p + 8);
  #pragma unroll
  for (int e = 0; e < 8; ++e) { a[e] = r0[e]; a[8 + e] = r1[e]; }
  return a;
}

// ---- B fragment: col = colBase+lane%16, K run [kb,kb+16), kb=kk+(lane/16)*16
template<bool VEC>
DEVFN v16h fragB_any(const float* __restrict__ p, long ks)
{
  v16h b;
  if (VEC) {
    const float4 x0 = *(const float4*)(p);
    const float4 x1 = *(const float4*)(p + 4);
    const float4 x2 = *(const float4*)(p + 8);
    const float4 x3 = *(const float4*)(p + 12);
    b[0]=(_Float16)x0.x; b[1]=(_Float16)x0.y; b[2]=(_Float16)x0.z; b[3]=(_Float16)x0.w;
    b[4]=(_Float16)x1.x; b[5]=(_Float16)x1.y; b[6]=(_Float16)x1.z; b[7]=(_Float16)x1.w;
    b[8]=(_Float16)x2.x; b[9]=(_Float16)x2.y; b[10]=(_Float16)x2.z; b[11]=(_Float16)x2.w;
    b[12]=(_Float16)x3.x; b[13]=(_Float16)x3.y; b[14]=(_Float16)x3.z; b[15]=(_Float16)x3.w;
  } else {
    #pragma unroll
    for (int e = 0; e < 16; ++e) b[e] = (_Float16)p[(long)e * ks];
  }
  return b;
}
template<bool VEC>
DEVFN v16h fragB_any(const _Float16* __restrict__ p, long ks)
{
  v16h b;
  if (VEC) {
    const h8 r0 = *(const h8*)(p);
    const h8 r1 = *(const h8*)(p + 8);
    #pragma unroll
    for (int e = 0; e < 8; ++e) { b[e] = r0[e]; b[8 + e] = r1[e]; }
  } else {
    #pragma unroll
    for (int e = 0; e < 16; ++e) b[e] = p[(long)e * ks];
  }
  return b;
}

// tmode: 0 -> idx = row*crs + col                              (normal)
//        1 -> idx = (row>>8)*16384 + col*256 + (row&255)       (VT1 store)
//        2 -> ((row>>16)*256 + (row&255))*16384 + col*256 + ((row>>8)&255)
template<typename CT>
DEVFN void epilogue(CT* __restrict__ C, const v8f& acc, int m0, int col,
                    long crs, float bv, int act, int beta, int half, int tmode)
{
  #pragma unroll
  for (int r = 0; r < 8; ++r) {
    const int row = m0 + r + half * 8;
    long idx;
    if      (tmode == 1) idx = (long)(row >> 8) * 16384 + (long)col * 256 + (row & 255);
    else if (tmode == 2) idx = ((long)(row >> 16) * 256 + (row & 255)) * 16384
                               + (long)col * 256 + ((row >> 8) & 255);
    else                 idx = (long)row * crs + col;
    float v = acc[r] + bv;
    if (beta) v += (float)C[idx];
    if      (act == 1) v = fmaxf(v, 0.f);
    else if (act == 2) v = sigm(v);
    else if (act == 3) v = lrelu(v);
    C[idx] = (CT)v;
  }
}

// ---------------------------------------------------------------------------
// Generic batched GEMM: C[m,n] = act( sum_k A[m,k] * W[n,k] + bias[n] (+C) )
// Batch index z: operand base += (z/inner)*b1 + (z%inner)*b2 (element strides)
// A has unit k-stride (always true in this net). K multiple of 32.
// ---------------------------------------------------------------------------
template<typename AT, typename BT, typename CT, bool BV>
__global__ __launch_bounds__(128)
void gemmk(const AT* __restrict__ A, long ab1, long ab2, long ars,
           const BT* __restrict__ W, long wb1, long wb2, long wns, long wks,
           const float* __restrict__ bias,
           CT* __restrict__ C, long cb1, long cb2, long crs,
           int M, int Nout, int K, int act, int beta, int inner, int tmode)
{
  const int lane = threadIdx.x & 31;
  const int wv   = threadIdx.x >> 5;
  const int m0   = (blockIdx.x * 4 + wv) * 32;
  if (m0 >= M) return;
  const int n0   = blockIdx.y * 32;
  const int z    = blockIdx.z;
  A += (long)(z / inner) * ab1 + (long)(z % inner) * ab2;
  W += (long)(z / inner) * wb1 + (long)(z % inner) * wb2;
  C += (long)(z / inner) * cb1 + (long)(z % inner) * cb2;

  const int half = lane >> 4;   // 0 or 1
  const int l16  = lane & 15;

  const AT* pa0 = A + (long)(m0 + l16)      * ars + (long)(half * 16);
  const AT* pa1 = A + (long)(m0 + 16 + l16) * ars + (long)(half * 16);
  const BT* pb0 = W + (long)(n0 + l16)      * wns + (long)(half * 16) * wks;
  const BT* pb1 = W + (long)(n0 + 16 + l16) * wns + (long)(half * 16) * wks;

  v8f acc00 = {}, acc01 = {}, acc10 = {}, acc11 = {};
  for (int kk = 0; kk < K; kk += 32) {
    const v16h a0 = fragA_any(pa0);
    const v16h a1 = fragA_any(pa1);
    const v16h b0 = fragB_any<BV>(pb0, wks);
    const v16h b1 = fragB_any<BV>(pb1, wks);
    if (kk + 32 < K) {
      __builtin_prefetch(pa0 + 32, 0, 1);     // global_prefetch next K slab
      __builtin_prefetch(pa1 + 32, 0, 1);
    }
    acc00 = __builtin_amdgcn_wmma_f32_16x16x32_f16(false, a0, false, b0, (short)0, acc00, false, false);
    acc01 = __builtin_amdgcn_wmma_f32_16x16x32_f16(false, a0, false, b1, (short)0, acc01, false, false);
    acc10 = __builtin_amdgcn_wmma_f32_16x16x32_f16(false, a1, false, b0, (short)0, acc10, false, false);
    acc11 = __builtin_amdgcn_wmma_f32_16x16x32_f16(false, a1, false, b1, (short)0, acc11, false, false);
    pa0 += 32;        pa1 += 32;
    pb0 += 32 * wks;  pb1 += 32 * wks;
  }

  const int   c0  = n0 + l16;
  const int   c1  = n0 + 16 + l16;
  const float bv0 = bias ? bias[c0] : 0.f;
  const float bv1 = bias ? bias[c1] : 0.f;
  epilogue(C, acc00, m0,      c0, crs, bv0, act, beta, half, tmode);
  epilogue(C, acc01, m0,      c1, crs, bv1, act, beta, half, tmode);
  epilogue(C, acc10, m0 + 16, c0, crs, bv0, act, beta, half, tmode);
  epilogue(C, acc11, m0 + 16, c1, crs, bv1, act, beta, half, tmode);
}

// ---- LayerNorm over 64-wide rows, one wave per row --------------------------
__global__ void ln64(const float* __restrict__ x, const float* __restrict__ g,
                     const float* __restrict__ b, float* __restrict__ y, long M)
{
  long row = (long)blockIdx.x * 8 + (threadIdx.x >> 5);
  if (row >= M) return;
  int lane = threadIdx.x & 31;
  const float* xr = x + row * FD;
  float a0 = xr[lane], a1 = xr[lane + 32];
  float s  = wsum(a0 + a1);
  float s2 = wsum(a0 * a0 + a1 * a1);
  float mean = s * (1.f / 64.f);
  float var  = s2 * (1.f / 64.f) - mean * mean;
  float inv  = rsqrtf(var + 1e-5f);
  y[row * FD + lane]      = (a0 - mean) * inv * g[lane]      + b[lane];
  y[row * FD + lane + 32] = (a1 - mean) * inv * g[lane + 32] + b[lane + 32];
}

// ---- generic 64-wide dot per row -------------------------------------------
__global__ void dot64(const float* __restrict__ X, const float* __restrict__ w,
                      const float* __restrict__ wb, float* __restrict__ out, long M)
{
  long row = (long)blockIdx.x * 8 + (threadIdx.x >> 5);
  if (row >= M) return;
  int lane = threadIdx.x & 31;
  const float* xr = X + row * FD;
  float s = wsum(xr[lane] * w[lane] + xr[lane + 32] * w[lane + 32]);
  if (lane == 0) out[row] = s + (wb ? wb[0] : 0.f);
}

// ---- NodeInteraction gate + o matrix (third output) -------------------------
__global__ void gate_o(const float* __restrict__ E2, const float* __restrict__ Ww,
                       const float* __restrict__ Wb, const float* __restrict__ adj,
                       float* __restrict__ o, long Me)
{
  long e = (long)blockIdx.x * 8 + (threadIdx.x >> 5);
  if (e >= Me) return;
  int lane = threadIdx.x & 31;
  long b = e >> 16;
  int ij = (int)(e & 65535);
  int i = ij >> 8, j = ij & 255;
  const float* r1 = E2 + e * FD;
  const float* r2 = E2 + ((b << 16) + ((long)j << 8) + i) * FD;
  float s = r1[lane] * Ww[lane]           + r1[lane + 32] * Ww[lane + 32]
          + r2[lane] * Ww[FD + lane]      + r2[lane + 32] * Ww[FD + 32 + lane];
  s = wsum(s);
  if (lane == 0) {
    float gv = sigm(s + Wb[0]);
    o[e] = (adj[e] + gv) * gv;
  }
}

// ---- GAT: e1/e2h dot products ----------------------------------------------
__global__ void gat_ev(const float* __restrict__ Hb, const float* __restrict__ gatA,
                       float* __restrict__ e1, float* __restrict__ e2h)
{
  int z = blockIdx.x * 8 + (threadIdx.x >> 5);      // (b*H+h)*N + n
  if (z >= NB * NH * NN_) return;
  int lane = threadIdx.x & 31;
  int hh = (z >> 8) & (NH - 1);
  const float* xr = Hb + (long)z * FD;
  const float* A  = gatA + hh * 2 * FD;
  float x0 = xr[lane], x1 = xr[lane + 32];
  float s1 = wsum(x0 * A[lane]       + x1 * A[lane + 32]);
  float s2 = wsum(x0 * A[FD + lane]  + x1 * A[FD + 32 + lane]);
  if (lane == 0) { e1[z] = s1; e2h[z] = s2; }
}

// ---- GAT attention softmax (width 256, one wave per row) --------------------
__global__ void gat_att(const float* __restrict__ e1, const float* __restrict__ e2h,
                        const float* __restrict__ o, float* __restrict__ att)
{
  int z = blockIdx.x * 8 + (threadIdx.x >> 5);      // (b*H+h)*N + i
  if (z >= NB * NH * NN_) return;
  int lane = threadIdx.x & 31;
  int i  = z & 255;
  int bh = z >> 8;
  int b  = bh >> 2;
  const float* e2b = e2h + ((long)bh << 8);
  const float* ob  = o + ((long)b << 16) + ((long)i << 8);
  float ei = e1[z];
  float v[8]; float mx = -1e30f;
  #pragma unroll
  for (int t = 0; t < 8; ++t) {
    int j = lane + t * 32;
    float s = lrelu(ei + e2b[j]) * ob[j];
    v[t] = s; mx = fmaxf(mx, s);
  }
  mx = wmax(mx);
  float s = 0.f;
  #pragma unroll
  for (int t = 0; t < 8; ++t) { v[t] = __expf(v[t] - mx); s += v[t]; }
  s = wsum(s);
  float inv = 1.f / s;
  #pragma unroll
  for (int t = 0; t < 8; ++t) att[(long)z * NN_ + lane + t * 32] = v[t] * inv;
}

// ---- concat heads: cat[b,n,h*64+d] = hp[b,h,n,d] ---------------------------
__global__ void cat_k(const float* __restrict__ HP, float* __restrict__ CAT)
{
  int idx = blockIdx.x * 256 + threadIdx.x;         // < 131072
  int b = idx >> 16;
  int rem = idx & 65535;
  int n = rem >> 8;
  int c = rem & 255;
  int h = c >> 6, d = c & 63;
  CAT[idx] = HP[(((long)(b * NH + h) * NN_) + n) * FD + d];
}

// ---- EdgeTransformer score softmax -> f16 probabilities --------------------
__global__ void et_softmax(const float* __restrict__ A, const float* __restrict__ bm,
                           _Float16* __restrict__ outH, long Me)
{
  long r = (long)blockIdx.x * 8 + (threadIdx.x >> 5);
  if (r >= Me) return;
  int lane = threadIdx.x & 31;
  long b = r >> 16;
  int ij = (int)(r & 65535);
  int i = ij >> 8;
  const float* row = A + r * NN_;
  const float* bmb = bm + (b << 16);
  float v[8]; float mx = -1e30f;
  #pragma unroll
  for (int t = 0; t < 8; ++t) {
    int l = lane + t * 32;
    float s = row[l] * 0.125f + bmb[(long)i * NN_ + l] + bmb[(long)l * NN_ + i];
    v[t] = s; mx = fmaxf(mx, s);
  }
  mx = wmax(mx);
  float s = 0.f;
  #pragma unroll
  for (int t = 0; t < 8; ++t) { v[t] = __expf(v[t] - mx); s += v[t]; }
  s = wsum(s);
  float inv = 1.f / s;
  #pragma unroll
  for (int t = 0; t < 8; ++t)
    outH[r * NN_ + lane + t * 32] = (_Float16)(v[t] * inv);
}

// ---- elementwise multiply (gate) -------------------------------------------
__global__ void gmul(float* __restrict__ X, const float* __restrict__ G, long total)
{
  long i = (long)blockIdx.x * 256 + threadIdx.x;
  if (i < total) X[i] *= G[i];
}

// ---- NodeUpdate / EdgeUpdate softmax: arg_j = rb[z] + Wm[b,(i,j) or (j,i)] --
__global__ void softmax_bias(const float* __restrict__ rb, const float* __restrict__ Wm,
                             float* __restrict__ outW, int trans)
{
  int z = blockIdx.x * 8 + (threadIdx.x >> 5);      // b*N + i
  if (z >= NB * NN_) return;
  int lane = threadIdx.x & 31;
  int b = z >> 8, i = z & 255;
  long base = (long)b << 16;
  float r = rb[z];
  float v[8]; float mx = -1e30f;
  #pragma unroll
  for (int t = 0; t < 8; ++t) {
    int j = lane + t * 32;
    long idx = trans ? base + (long)j * NN_ + i : base + (long)i * NN_ + j;
    v[t] = r + Wm[idx]; mx = fmaxf(mx, v[t]);
  }
  mx = wmax(mx);
  float s = 0.f;
  #pragma unroll
  for (int t = 0; t < 8; ++t) { v[t] = __expf(v[t] - mx); s += v[t]; }
  s = wsum(s);
  float inv = 1.f / s;
  #pragma unroll
  for (int t = 0; t < 8; ++t) outW[base + (long)i * NN_ + lane + t * 32] = v[t] * inv;
}

// ---- NodeUpdate weighted sum: node_out[b,j,d] ------------------------------
__global__ void nu_sum(const float* __restrict__ WF, const float* __restrict__ QN,
                       const float* __restrict__ N3, float* __restrict__ outN)
{
  int z = blockIdx.x;           // b*N + j
  int d = threadIdx.x;          // 0..63
  int b = z >> 8, j = z & 255;
  long base = (long)b << 16;
  const float* wrow = WF + base + (long)j * NN_;
  float acc = 0.f;
  for (int i = 0; i < NN_; ++i)
    acc += wrow[i] * QN[(base + (long)i * NN_ + j) * FD + d];
  outN[(long)z * FD + d] = fmaxf(acc, 0.f) + N3[(long)z * FD + d];
}

// ---- EdgeUpdate final combine ----------------------------------------------
__global__ void edge_out_k(const float* __restrict__ WE, const float* __restrict__ QE,
                           const float* __restrict__ ED2, float* __restrict__ out, long total)
{
  long idx = (long)blockIdx.x * 256 + threadIdx.x;
  if (idx >= total) return;
  long e = idx >> 6; int d = (int)(idx & 63);
  long b = e >> 16;  int i = (int)((e >> 8) & 255);
  float v = fmaxf(WE[e] * QE[((b << 8) + i) * FD + d], 0.f) + ED2[idx];
  out[idx] = v;
}

// ---------------------------------------------------------------------------
template<typename AT, typename BT, typename CT>
static void gemmT(hipStream_t s,
                  const AT* A, long ab1, long ab2, long ars,
                  const BT* W, long wb1, long wb2, long wns, long wks,
                  const float* bias, CT* C, long cb1, long cb2, long crs,
                  int M, int Nout, int K, int act, int beta, int batch, int inner,
                  int tmode = 0)
{
  dim3 g((M + 127) / 128, Nout / 32, batch), bl(128);
  if (wks == 1)
    gemmk<AT, BT, CT, true><<<g, bl, 0, s>>>(A, ab1, ab2, ars, W, wb1, wb2, wns, wks,
                                             bias, C, cb1, cb2, crs, M, Nout, K, act, beta, inner, tmode);
  else
    gemmk<AT, BT, CT, false><<<g, bl, 0, s>>>(A, ab1, ab2, ars, W, wb1, wb2, wns, wks,
                                              bias, C, cb1, cb2, crs, M, Nout, K, act, beta, inner, tmode);
}

extern "C" void kernel_launch(void* const* d_in, const int* in_sizes, int n_in,
                              void* d_out, int out_size, void* d_ws, size_t ws_size,
                              hipStream_t stream)
{
  (void)in_sizes; (void)n_in; (void)out_size; (void)ws_size;
  const float* nodes     = (const float*)d_in[0];
  const float* edges     = (const float*)d_in[1];
  const float* adj       = (const float*)d_in[2];
  const float* ln1_g     = (const float*)d_in[3];
  const float* ln1_b     = (const float*)d_in[4];
  const float* ln2_g     = (const float*)d_in[5];
  const float* ln2_b     = (const float*)d_in[6];
  const float* ni_Win_w  = (const float*)d_in[7];
  const float* ni_Win_b  = (const float*)d_in[8];
  const float* ni_W_w    = (const float*)d_in[9];
  const float* ni_W_b    = (const float*)d_in[10];
  const float* gat_W     = (const float*)d_in[11];
  const float* gat_A     = (const float*)d_in[12];
  const float* gat_out_w = (const float*)d_in[13];
  const float* gat_out_b = (const float*)d_in[14];
  const float* et_Wq_w   = (const float*)d_in[15];
  const float* et_Wq_b   = (const float*)d_in[16];
  const float* et_Wk_w   = (const float*)d_in[17];
  const float* et_Wk_b   = (const float*)d_in[18];
  const float* et_Wv_w   = (const float*)d_in[19];
  const float* et_Wv_b   = (const float*)d_in[20];
  const float* et_Wg2_w  = (const float*)d_in[21];
  const float* et_Wg2_b  = (const float*)d_in[22];
  const float* et_Wo_w   = (const float*)d_in[23];
  const float* et_Wo_b   = (const float*)d_in[24];
  const float* et_Wb_w   = (const float*)d_in[25];
  const float* et_Wb_b   = (const float*)d_in[26];
  const float* f1_ln_g   = (const float*)d_in[27];
  const float* f1_ln_b   = (const float*)d_in[28];
  const float* f1_w1     = (const float*)d_in[29];
  const float* f1_b1     = (const float*)d_in[30];
  const float* f1_w2     = (const float*)d_in[31];
  const float* f1_b2     = (const float*)d_in[32];
  const float* f2_ln_g   = (const float*)d_in[33];
  const float* f2_ln_b   = (const float*)d_in[34];
  const float* f2_w1     = (const float*)d_in[35];
  const float* f2_b1     = (const float*)d_in[36];
  const float* f2_w2     = (const float*)d_in[37];
  const float* f2_b2     = (const float*)d_in[38];
  const float* nu_Q      = (const float*)d_in[39];
  const float* nu_A      = (const float*)d_in[40];
  const float* eu_Q      = (const float*)d_in[41];
  const float* eu_A      = (const float*)d_in[42];

  float* out_node = (float*)d_out;                      // [2,256,64]
  float* out_edge = out_node + (long)NB * NN_ * FD;     // [2,65536,64]
  float* out_o    = out_edge + ME * FD;                 // [2,256,256]

  // ---- workspace layout (floats) ----
  float*     ws  = (float*)d_ws;
  const long BIG = ME * FD;                             // 8388608
  float* EN = ws;               // edges_n; later AH(lo half), then FNN1-LN out
  float* E2 = ws + 1 * BIG;     // NI e2;   later AH(hi half)
  float* Qb = ws + 2 * BIG;     // ET q;    later O12 = o1+o2
  float* Kb = ws + 3 * BIG;     // ET k;    later edges2
  float* Vb = ws + 4 * BIG;     // VT1+VT2 (f16); later qn (f32)
  float* Gb = ws + 5 * BIG;     // ET gate; later ET output
  float* Ab = ws + 6 * BIG;     // f32 scores [2,256,256,256]; later HH (f16)
  float* BM = ws + 10 * BIG;    // ET bias vector [2,65536]
  _Float16* AH  = (_Float16*)EN;   // f16 attention probs [ME,256] (EN+E2)
  _Float16* HH  = (_Float16*)Ab;   // f16 FNN1 hidden [ME,256]
  _Float16* VT1 = (_Float16*)Vb;   // f16 V^T per (b,i): [b,i,d,l]
  _Float16* VT2 = VT1 + BIG;       // f16 V^T per (b,i) of v[b,l,i,d]
  float* sm = BM + ME;
  float* NNb = sm; sm += (long)NB * NN_ * FD;           // nodes_n
  float* Hb  = sm; sm += (long)NB * NH * NN_ * FD;      // GAT h
  float* E1  = sm; sm += (long)NB * NH * NN_;
  float* E2H = sm; sm += (long)NB * NH * NN_;
  float* ATT = sm; sm += (long)NB * NH * NN_ * NN_;
  float* HP  = sm; sm += (long)NB * NH * NN_ * FD;
  float* CAT = sm; sm += (long)NB * NN_ * NH * FD;
  float* N2  = sm; sm += (long)NB * NN_ * FD;           // nodes2
  float* N2L = sm; sm += (long)NB * NN_ * FD;
  float* FH2 = sm; sm += (long)NB * NN_ * 4 * FD;
  float* N3  = sm; sm += (long)NB * NN_ * FD;           // nodes3
  float* W1N = sm; sm += (long)NB * NN_;
  float* W2N = sm; sm += ME;
  float* WF  = sm; sm += ME;
  float* QE  = sm; sm += (long)NB * NN_ * FD;
  float* WE1 = sm; sm += (long)NB * NN_;
  float* WE2 = sm; sm += ME;
  float* WEb = sm; sm += ME;

  const long N2_ = (long)NN_ * NN_;                     // 65536
  const long N3_ = N2_ * NN_;                           // 16777216
  const int  MEi = (int)ME;

  // 1) LayerNorms of inputs
  ln64<<<(MEi + 7) / 8, 256, 0, stream>>>(edges, ln1_g, ln1_b, EN, ME);
  ln64<<<64, 256, 0, stream>>>(nodes, ln2_g, ln2_b, NNb, NB * NN_);

  // 2) NodeInteraction: e2 = relu(EN @ Win^T + b); gate -> o (3rd output)
  gemmT(stream, EN,0,0,64, ni_Win_w,0,0,64,1, ni_Win_b, E2,0,0,64, MEi,64,64, 1,0, 1,1);
  gate_o<<<(MEi + 7) / 8, 256, 0, stream>>>(E2, ni_W_w, ni_W_b, adj, out_o, ME);

  // 3) GAT
  gemmT(stream, NNb, (long)NN_*FD,0, 64, gat_W, 0,(long)FD*FD, 64,1, nullptr,
        Hb, (long)NH*NN_*FD,(long)NN_*FD, 64, NN_,64,64, 0,0, NB*NH, NH);
  gat_ev<<<(NB*NH*NN_ + 7) / 8, 256, 0, stream>>>(Hb, gat_A, E1, E2H);
  gat_att<<<(NB*NH*NN_ + 7) / 8, 256, 0, stream>>>(E1, E2H, out_o, ATT);
  gemmT(stream, ATT, N2_,0, NN_, Hb, (long)NN_*FD,0, 1,64, nullptr,
        HP, (long)NN_*FD,0, 64, NN_,64,NN_, 3,0, NB*NH, 1);
  cat_k<<<512, 256, 0, stream>>>(HP, CAT);
  gemmT(stream, CAT,0,0,256, gat_out_w,0,0,256,1, gat_out_b, N2,0,0,64,
        NB*NN_,64,256, 3,0, 1,1);

  // 4) EdgeTransformer projections: q, k, gate, bias; V directly into two
  //    transposed f16 layouts (tmode 1/2) so o1/o2 stream b128 B fragments.
  gemmT(stream, EN,0,0,64, et_Wq_w,0,0,64,1, et_Wq_b, Qb,0,0,64, MEi,64,64, 0,0,1,1);
  gemmT(stream, EN,0,0,64, et_Wk_w,0,0,64,1, et_Wk_b, Kb,0,0,64, MEi,64,64, 0,0,1,1);
  gemmT(stream, EN,0,0,64, et_Wv_w,0,0,64,1, et_Wv_b, VT1,0,0,0, MEi,64,64, 0,0,1,1, 1);
  gemmT(stream, EN,0,0,64, et_Wv_w,0,0,64,1, et_Wv_b, VT2,0,0,0, MEi,64,64, 0,0,1,1, 2);
  gemmT(stream, EN,0,0,64, et_Wg2_w,0,0,64,1, et_Wg2_b, Gb,0,0,64, MEi,64,64, 2,0,1,1);
  dot64<<<(MEi + 7) / 8, 256, 0, stream>>>(EN, et_Wb_w, et_Wb_b, BM, ME);

  // 5) Scores: t1 per (b,i); t2 per (b,j) accumulated; softmax -> f16 AH
  gemmT(stream, Qb, (long)NN_*FD,0, 64, Kb, (long)NN_*FD,0, 64,1, nullptr,
        Ab, N2_,0, NN_, NN_,NN_,64, 0,0, NB*NN_, 1);
  gemmT(stream, Qb, N2_*FD,64, (long)NN_*FD, Kb, N2_*FD,64, (long)NN_*FD,1, nullptr,
        Ab, N3_,NN_, N2_, NN_,NN_,64, 0,1, NB*NN_, NN_);
  et_softmax<<<(MEi + 7) / 8, 256, 0, stream>>>(Ab, BM, AH, ME);

  // 6) o1/o2: f16 probs x f16 V^T -> O12 in Qb; gate; output projection
  gemmT(stream, (const _Float16*)AH, N2_,0, NN_,
        (const _Float16*)VT1, (long)FD*NN_,0, NN_,1, nullptr,
        Qb, (long)NN_*FD,0, 64, NN_,64,NN_, 0,0, NB*NN_, 1);
  gemmT(stream, (const _Float16*)AH, N3_,NN_, N2_,
        (const _Float16*)VT2, (long)FD*NN_*NN_,(long)FD*NN_, NN_,1, nullptr,
        Qb, N2_*FD,(long)NN_*FD, 64, NN_,64,NN_, 0,1, NB*NN_, NN_);
  gmul<<<(unsigned)((ME*FD) / 256), 256, 0, stream>>>(Qb, Gb, ME * FD);
  gemmT(stream, Qb,0,0,64, et_Wo_w,0,0,64,1, et_Wo_b, Gb,0,0,64, MEi,64,64, 1,0,1,1);

  // 7) FNN1 (edges): LN -> 64->256 relu (f16 hidden) -> 256->64 relu
  ln64<<<(MEi + 7) / 8, 256, 0, stream>>>(Gb, f1_ln_g, f1_ln_b, EN, ME);
  gemmT(stream, EN,0,0,64, f1_w1,0,0,64,1, f1_b1, HH,0,0,256, MEi,256,64, 1,0,1,1);
  gemmT(stream, (const _Float16*)HH,0,0,256, f1_w2,0,0,256,1, f1_b2, Kb,0,0,64,
        MEi,64,256, 1,0,1,1);

  // 8) FNN2 (nodes)
  ln64<<<64, 256, 0, stream>>>(N2, f2_ln_g, f2_ln_b, N2L, NB * NN_);
  gemmT(stream, N2L,0,0,64, f2_w1,0,0,64,1, f2_b1, FH2,0,0,256, NB*NN_,256,64, 1,0,1,1);
  gemmT(stream, FH2,0,0,256, f2_w2,0,0,256,1, f2_b2, N3,0,0,64, NB*NN_,64,256, 1,0,1,1);

  // 9) NodeUpdate
  gemmT(stream, Kb,0,0,64, nu_Q,0,0,64,1, nullptr, Vb,0,0,64, MEi,64,64, 0,0,1,1);
  dot64<<<64, 256, 0, stream>>>(N3, nu_A, nullptr, W1N, NB * NN_);
  dot64<<<(MEi + 7) / 8, 256, 0, stream>>>(Vb, nu_A + 64, nullptr, W2N, ME);
  softmax_bias<<<64, 256, 0, stream>>>(W1N, W2N, WF, 1);
  nu_sum<<<NB * NN_, FD, 0, stream>>>(WF, Vb, N3, out_node);

  // 10) EdgeUpdate
  gemmT(stream, N3,0,0,64, eu_Q,0,0,64,1, nullptr, QE,0,0,64, NB*NN_,64,64, 0,0,1,1);
  dot64<<<64, 256, 0, stream>>>(QE, eu_A, nullptr, WE1, NB * NN_);
  dot64<<<(MEi + 7) / 8, 256, 0, stream>>>(Kb, eu_A + 64, nullptr, WE2, ME);
  softmax_bias<<<64, 256, 0, stream>>>(WE1, WE2, WEb, 0);
  edge_out_k<<<(unsigned)((ME*FD) / 256), 256, 0, stream>>>(WEb, QE, Kb, out_edge, ME * FD);
}
